// NonLocalBlock_67611375173833
// MI455X (gfx1250) — matrix operand order repaired and checked by reading, not compiled, not run on hardware
//
#include <hip/hip_runtime.h>
#include <hip/hip_bf16.h>
#include <math.h>

// ---------------- problem constants ----------------
#define B_   4
#define C_   256
#define C2_  128
#define HIN  128
#define HO   63
#define NN   3969          // HO*HO
#define NP   4032          // padded to 63 tiles of 64

typedef unsigned short u16;
typedef unsigned int   u32;
typedef __attribute__((ext_vector_type(16))) __bf16 v16bf;
typedef __attribute__((ext_vector_type(8)))  float  v8f;

union AF { v16bf v; u16 u[16]; u32 d[8]; uint4 q[2]; };

// native f32 -> bf16 (RNE): lowers to v_cvt_pk_bf16_f32 (confirmed in round-3 asm)
__device__ __forceinline__ u16 f2bf(float f) {
    union { __bf16 b; u16 u; } x;
    x.b = (__bf16)f;
    return x.u;
}
__device__ __forceinline__ u32 f2bf2(float a, float b) {
    return (u32)f2bf(a) | ((u32)f2bf(b) << 16);
}
// A-fragment: lane holds row m; k = khalf*16 + j contiguous
__device__ __forceinline__ AF load_afrag(const u16* rowp, int khalf) {
    AF f;
    f.q[0] = *(const uint4*)(rowp + khalf * 16);
    f.q[1] = *(const uint4*)(rowp + khalf * 16 + 8);
    return f;
}
// B-fragment: lane holds column n; k = khalf*16 + j contiguous (p already at that base)
__device__ __forceinline__ AF load_bfrag(const u16* p) {
    AF f;
    f.q[0] = *(const uint4*)p;
    f.q[1] = *(const uint4*)(p + 8);
    return f;
}
// async global -> LDS 16B copy; OFF advances both global and LDS address (ISA 15.18.3)
template<int OFF>
__device__ __forceinline__ void async_g2l(u32 lds_addr, const void* gptr) {
    asm volatile("global_load_async_to_lds_b128 %0, %1, off offset:%2"
                 :: "v"(lds_addr), "v"(gptr), "n"(OFF) : "memory");
}
__device__ __forceinline__ void async_row128(u32 lds_addr, const void* gptr) {
    async_g2l<0>(lds_addr, gptr);   async_g2l<16>(lds_addr, gptr);
    async_g2l<32>(lds_addr, gptr);  async_g2l<48>(lds_addr, gptr);
    async_g2l<64>(lds_addr, gptr);  async_g2l<80>(lds_addr, gptr);
    async_g2l<96>(lds_addr, gptr);  async_g2l<112>(lds_addr, gptr);
}

// ---------------- kernel 1: maxpool 3x3 s2 + bf16 transpose to position-major ----------------
// resid: (B,C,NP) fp32 row-major (n contiguous).  xpbT: (B,NP,C) bf16 (c contiguous).
__global__ void __launch_bounds__(256) k_pool(const float* __restrict__ x,
                                              float* __restrict__ resid, u16* __restrict__ xpbT) {
    __shared__ __align__(16) u16 T[64 * 72];      // [n within tile][c within tile]
    const int t = threadIdx.x;
    const int n0 = blockIdx.x * 64, c0 = blockIdx.y * 64, b = blockIdx.z;
    const int ni = t & 63, ci = t >> 6;           // thread: position n0+ni, channels c0+ci*16..+15
    const int n = n0 + ni;
    #pragma unroll 4
    for (int cc = 0; cc < 16; ++cc) {
        const int c = c0 + ci * 16 + cc;
        float m = 0.f;
        if (n < NN) {
            int h = n / HO, ww = n % HO;
            const float* p = x + ((size_t)(b * C_ + c)) * HIN * HIN + (2 * h) * HIN + 2 * ww;
            m = -INFINITY;
            #pragma unroll
            for (int i = 0; i < 3; ++i)
                #pragma unroll
                for (int j = 0; j < 3; ++j) m = fmaxf(m, p[i * HIN + j]);
        }
        resid[((size_t)(b * C_ + c)) * NP + n] = m;
        T[ni * 72 + ci * 16 + cc] = f2bf(m);
    }
    __syncthreads();
    { // write xpbT coalesced: 64 rows (n) x 64 c, 16 halfs per thread
        const int row = t >> 2, qq = t & 3;
        u16* dst = xpbT + (size_t)b * NP * C_ + (size_t)(n0 + row) * C_ + c0 + qq * 16;
        const u16* src = T + row * 72 + qq * 16;
        *(uint4*)(dst)     = *(const uint4*)(src);
        *(uint4*)(dst + 8) = *(const uint4*)(src + 8);
    }
}

// ---------------- kernel: zero BN stats ----------------
__global__ void k_zero(float* stats) { stats[threadIdx.x] = 0.f; }

// ---------------- kernel 2: projections  W(128x256) @ x(256xNP) ----------------
// mode 0 -> thetaT (NP x C2), mode 1 -> phiT (NP x C2), mode 2 -> gC (C2 x NP)
__global__ void __launch_bounds__(128) k_proj(const float* __restrict__ thw, const float* __restrict__ phw,
                                              const float* __restrict__ gw,  const u16* __restrict__ xpbT,
                                              u16* __restrict__ thT, u16* __restrict__ phiT, u16* __restrict__ gC) {
    __shared__ __align__(16) u16 As[64 * 72];     // [m][k]  (weights, row-major)
    __shared__ __align__(16) u16 Bs[64 * 72];     // [n][k]  (activations, k-contiguous)
    const int t = threadIdx.x, lane = t & 31, w = t >> 5;
    const int ntile = blockIdx.x, mtile = blockIdx.y, z = blockIdx.z;
    const int mode = z % 3, b = z / 3;
    const float* W = (mode == 0) ? thw : (mode == 1) ? phw : gw;
    const int n0 = ntile * 64, m0 = mtile * 64;
    const int nc = lane & 15, khalf = lane >> 4, md = khalf * 8;
    v8f acc[4] = {};
    for (int kc = 0; kc < C_; kc += 64) {
        { // A chunk: W rows m0..m0+63, cols kc..kc+63 (fp32 -> bf16, packed dword stores)
            int row = t >> 1, hs = t & 1;
            const float* src = W + (size_t)(m0 + row) * C_ + kc + hs * 32;
            u16* dst = As + row * 72 + hs * 32;
            #pragma unroll
            for (int j = 0; j < 32; j += 4) {
                float4 f = *(const float4*)(src + j);
                *(u32*)(dst + j)     = f2bf2(f.x, f.y);
                *(u32*)(dst + j + 2) = f2bf2(f.z, f.w);
            }
        }
        { // B chunk: xpbT rows n0..n0+63, channels kc..kc+63 (16B coalesced, no transpose needed)
            int row = t >> 1, hs = t & 1;
            const u16* src = xpbT + (size_t)b * NP * C_ + (size_t)(n0 + row) * C_ + kc + hs * 32;
            u16* dst = Bs + row * 72 + hs * 32;
            #pragma unroll
            for (int j = 0; j < 32; j += 8) *(uint4*)(dst + j) = *(const uint4*)(src + j);
        }
        __syncthreads();
        #pragma unroll
        for (int ks = 0; ks < 64; ks += 32) {
            AF a = load_afrag(As + (w * 16 + nc) * 72 + ks, khalf);
            AF bb[4];
            #pragma unroll
            for (int ct = 0; ct < 4; ++ct)
                bb[ct] = load_bfrag(Bs + (ct * 16 + nc) * 72 + ks + khalf * 16);
            #pragma unroll
            for (int ct = 0; ct < 4; ++ct)
                acc[ct] = __builtin_amdgcn_wmma_f32_16x16x32_bf16(false, a.v, false, bb[ct].v,
                                                                  (short)0, acc[ct], false, false);
        }
        __syncthreads();
    }
    // store D tiles
    const int obase = m0 + w * 16 + md;           // even
    if (mode == 2) {
        #pragma unroll
        for (int ct = 0; ct < 4; ++ct)
            #pragma unroll
            for (int v = 0; v < 8; ++v)
                gC[(size_t)b * C2_ * NP + (size_t)(obase + v) * NP + n0 + ct * 16 + nc] = f2bf(acc[ct][v]);
    } else {
        u16* dst = (mode == 0) ? thT : phiT;
        #pragma unroll
        for (int ct = 0; ct < 4; ++ct) {
            size_t base = (size_t)b * NP * C2_ + (size_t)(n0 + ct * 16 + nc) * C2_ + obase;
            #pragma unroll
            for (int v = 0; v < 8; v += 2)
                *(u32*)(dst + base + v) = f2bf2(acc[ct][v], acc[ct][v + 1]);
        }
    }
}

// ---------------- kernel 3: flash attention  y = softmax(thetaT @ phi) @ g ----------------
__global__ void __launch_bounds__(128) k_attn(const u16* __restrict__ thT, const u16* __restrict__ phiT,
                                              const u16* __restrict__ gC, u16* __restrict__ ybuf) {
    __shared__ __align__(16) u16 phiS[64 * 136];  // [key][c]   (k-contiguous for S B-frags)
    __shared__ __align__(16) u16 gS[128 * 72];    // [c][key]   (k-contiguous for O B-frags)
    __shared__ __align__(16) u16 pS[4 * 16 * 72]; // per-wave P tile [q][key] (A-layout friendly)
    const int t = threadIdx.x, lane = t & 31, w = t >> 5;
    const int qt = blockIdx.x, b = blockIdx.y;
    const int q0w = qt * 64 + w * 16;
    const int nc = lane & 15, khalf = lane >> 4, md = khalf * 8;
    const size_t nB = (size_t)b * NP * C2_;       // base for thT / phiT / ybuf (position-major)
    const size_t cB = (size_t)b * C2_ * NP;       // base for gC (channel-major)

    // Q fragments (A-layout) straight from global: row q is k-contiguous
    AF qf[4];
    {
        const u16* qr = thT + nB + (size_t)(q0w + nc) * C2_;
        #pragma unroll
        for (int kk = 0; kk < 4; ++kk) qf[kk] = load_afrag(qr + kk * 32, khalf);
    }
    // per-thread async staging addresses (LDS dst fixed across tiles)
    const int pr = t >> 1, phs = t & 1;
    const u32 lds_phi = (u32)(uintptr_t)(phiS + pr * 136 + phs * 64);
    const u32 lds_g   = (u32)(uintptr_t)(gS + t * 72);

    float rmax[8], rsum[8];
    v8f O[8] = {};
    #pragma unroll
    for (int v = 0; v < 8; ++v) { rmax[v] = -INFINITY; rsum[v] = 0.f; }

    for (int kt = 0; kt < 63; ++kt) {
        const int k0 = kt * 64;
        // async stage phi tile (64 keys x 128 c) and g tile (128 c x 64 keys): VMEM -> LDS direct
        async_row128(lds_phi, phiT + nB + (size_t)(k0 + pr) * C2_ + phs * 64);
        async_row128(lds_g,   gC + cB + (size_t)t * NP + k0);
        if (kt + 1 < 63) __builtin_prefetch(gC + cB + (size_t)t * NP + k0 + 64, 0, 1);
        asm volatile("s_wait_asynccnt 0x0" ::: "memory");
        __syncthreads();

        // S = Q @ phi-tile : 16 WMMAs
        v8f S[4] = {};
        #pragma unroll
        for (int ks4 = 0; ks4 < 4; ++ks4) {
            AF bb[4];
            #pragma unroll
            for (int ct = 0; ct < 4; ++ct)
                bb[ct] = load_bfrag(phiS + (ct * 16 + nc) * 136 + ks4 * 32 + khalf * 16);
            #pragma unroll
            for (int ct = 0; ct < 4; ++ct)
                S[ct] = __builtin_amdgcn_wmma_f32_16x16x32_bf16(false, qf[ks4].v, false, bb[ct].v,
                                                                (short)0, S[ct], false, false);
        }
        bool kvalid[4];
        #pragma unroll
        for (int ct = 0; ct < 4; ++ct) kvalid[ct] = (k0 + ct * 16 + nc) < NN;

        // online softmax over 16-lane row groups of the D layout
        #pragma unroll
        for (int v = 0; v < 8; ++v) {
            float tm = -INFINITY;
            #pragma unroll
            for (int ct = 0; ct < 4; ++ct) if (kvalid[ct]) tm = fmaxf(tm, S[ct][v]);
            #pragma unroll
            for (int off = 8; off >= 1; off >>= 1) tm = fmaxf(tm, __shfl_xor(tm, off, 32));
            float nm = fmaxf(rmax[v], tm);
            float sc = __expf(rmax[v] - nm);
            rmax[v] = nm;
            float ps = 0.f;
            #pragma unroll
            for (int ct = 0; ct < 4; ++ct) {
                float p = kvalid[ct] ? __expf(S[ct][v] - nm) : 0.f;
                S[ct][v] = p; ps += p;
            }
            #pragma unroll
            for (int off = 8; off >= 1; off >>= 1) ps += __shfl_xor(ps, off, 32);
            rsum[v] = rsum[v] * sc + ps;
            #pragma unroll
            for (int ot = 0; ot < 8; ++ot) O[ot][v] *= sc;
        }
        // D-layout P -> per-wave LDS [q][key] (CDNA5 DS ops are in-order within a wave)
        u16* pw = pS + w * 16 * 72;
        #pragma unroll
        for (int ct = 0; ct < 4; ++ct)
            #pragma unroll
            for (int v = 0; v < 8; ++v)
                pw[(v + md) * 72 + ct * 16 + nc] = f2bf(S[ct][v]);
        asm volatile("s_wait_dscnt 0x0" ::: "memory");

        // O += P @ g-tile : 16 WMMAs
        #pragma unroll
        for (int ks2 = 0; ks2 < 2; ++ks2) {
            AF a = load_afrag(pw + nc * 72 + ks2 * 32, khalf);
            #pragma unroll
            for (int half = 0; half < 2; ++half) {
                AF bb[4];
                #pragma unroll
                for (int i = 0; i < 4; ++i)
                    bb[i] = load_bfrag(gS + ((half * 4 + i) * 16 + nc) * 72 + ks2 * 32 + khalf * 16);
                #pragma unroll
                for (int i = 0; i < 4; ++i)
                    O[half * 4 + i] = __builtin_amdgcn_wmma_f32_16x16x32_bf16(false, a.v, false, bb[i].v,
                                                                              (short)0, O[half * 4 + i],
                                                                              false, false);
            }
        }
        __syncthreads();
    }
    #pragma unroll
    for (int v = 0; v < 8; ++v) {
        int q = q0w + v + md;
        if (q < NN) {
            float inv = 1.f / rsum[v];
            #pragma unroll
            for (int ot = 0; ot < 8; ++ot)
                ybuf[nB + (size_t)q * C2_ + ot * 16 + nc] = f2bf(O[ot][v] * inv);
        }
    }
}

// ---------------- kernel 4: out_pre = y_w(256x128) @ raw-view(y) + BN partial stats ----------------
__global__ void __launch_bounds__(128) k_out(const float* __restrict__ yw, const u16* __restrict__ ybuf,
                                             float* __restrict__ outpre, float* __restrict__ stats) {
    __shared__ __align__(16) u16 As[64 * 136];    // [m][k] weights
    __shared__ __align__(16) u16 Bs[64 * 136];    // [n][k] raw-view Z, transposed at store time
    const int t = threadIdx.x, lane = t & 31, w = t >> 5;
    const int ntile = blockIdx.x, mtile = blockIdx.y, b = blockIdx.z;
    const int n0 = ntile * 64, m0 = mtile * 64;
    const int nc = lane & 15, khalf = lane >> 4, md = khalf * 8;
    { // A: y_w rows m0..m0+63 x 128
        int row = t >> 1, hs = t & 1;
        const float* src = yw + (size_t)(m0 + row) * C2_ + hs * 64;
        u16* dst = As + row * 136 + hs * 64;
        #pragma unroll
        for (int j = 0; j < 64; j += 4) {
            float4 f = *(const float4*)(src + j);
            *(u32*)(dst + j)     = f2bf2(f.x, f.y);
            *(u32*)(dst + j + 2) = f2bf2(f.z, f.w);
        }
    }
    { // B: Z[k][n] = y_flat[k*NN + n] (faithful torch .view, stride NN); transpose into [n][k]
        const u16* yb = ybuf + (size_t)b * NP * C2_;
        for (int j = 0; j < 64; ++j) {
            int n = n0 + j;
            Bs[j * 136 + t] = (n < NN) ? yb[(size_t)t * NN + n] : (u16)0;
        }
    }
    __syncthreads();
    v8f acc[4] = {};
    #pragma unroll
    for (int ks = 0; ks < 128; ks += 32) {
        AF a = load_afrag(As + (w * 16 + nc) * 136 + ks, khalf);
        AF bb[4];
        #pragma unroll
        for (int ct = 0; ct < 4; ++ct)
            bb[ct] = load_bfrag(Bs + (ct * 16 + nc) * 136 + ks + khalf * 16);
        #pragma unroll
        for (int ct = 0; ct < 4; ++ct)
            acc[ct] = __builtin_amdgcn_wmma_f32_16x16x32_bf16(false, a.v, false, bb[ct].v,
                                                              (short)0, acc[ct], false, false);
    }
    #pragma unroll
    for (int v = 0; v < 8; ++v) {
        int ch = m0 + w * 16 + v + md;
        float s = 0.f, sq = 0.f;
        #pragma unroll
        for (int ct = 0; ct < 4; ++ct) {
            int n = n0 + ct * 16 + nc;
            float val = acc[ct][v];
            if (n < NN) {
                outpre[(size_t)b * C_ * NP + (size_t)ch * NP + n] = val;
                s += val; sq += val * val;
            }
        }
        #pragma unroll
        for (int off = 8; off >= 1; off >>= 1) { s += __shfl_xor(s, off, 32); sq += __shfl_xor(sq, off, 32); }
        if (nc == 0) {
            atomicAdd(stats + ch, s);
            atomicAdd(stats + C_ + ch, sq);
        }
    }
}

// ---------------- kernel 5: BatchNorm(train) + residual + ReLU ----------------
__global__ void k_bnrelu(const float* __restrict__ outpre, const float* __restrict__ resid,
                         const float* __restrict__ stats, const float* __restrict__ bw,
                         const float* __restrict__ bb, float* __restrict__ out) {
    int idx = blockIdx.x * blockDim.x + threadIdx.x;
    if (idx >= B_ * C_ * NN) return;
    int n = idx % NN; int bc = idx / NN; int c = bc % C_;
    const float cnt = (float)(B_ * NN);
    float mean = stats[c] / cnt;
    float var  = stats[C_ + c] / cnt - mean * mean;
    size_t pidx = (size_t)bc * NP + n;
    float v = (outpre[pidx] - mean) * rsqrtf(var + 1e-5f) * bw[c] + bb[c] + resid[pidx];
    out[idx] = fmaxf(v, 0.f);
}

// ---------------- host launcher ----------------
extern "C" void kernel_launch(void* const* d_in, const int* in_sizes, int n_in,
                              void* d_out, int out_size, void* d_ws, size_t ws_size,
                              hipStream_t stream) {
    const float* x   = (const float*)d_in[0];
    const float* thw = (const float*)d_in[1];
    const float* phw = (const float*)d_in[2];
    const float* gw  = (const float*)d_in[3];
    const float* yw  = (const float*)d_in[4];
    const float* bnw = (const float*)d_in[5];
    const float* bnb = (const float*)d_in[6];

    // workspace layout (~57.8 MB total)
    constexpr size_t SZ_RESID = (size_t)B_ * C_ * NP * 4;
    constexpr size_t SZ_XPB   = (size_t)B_ * C_ * NP * 2;
    constexpr size_t SZ_PROJ  = (size_t)B_ * NP * C2_ * 2;
    constexpr size_t SZ_PRE   = (size_t)B_ * C_ * NP * 4;
    char* ws = (char*)d_ws;
    float* resid  = (float*)(ws);
    u16*   xpbT   = (u16*)(ws + SZ_RESID);
    u16*   thT    = (u16*)(ws + SZ_RESID + SZ_XPB);
    u16*   phiT   = (u16*)(ws + SZ_RESID + SZ_XPB + SZ_PROJ);
    u16*   gC     = (u16*)(ws + SZ_RESID + SZ_XPB + 2 * SZ_PROJ);
    u16*   ybuf   = (u16*)(ws + SZ_RESID + SZ_XPB + 3 * SZ_PROJ);
    float* outpre = (float*)(ws + SZ_RESID + SZ_XPB + 4 * SZ_PROJ);
    float* stats  = (float*)(ws + SZ_RESID + SZ_XPB + 4 * SZ_PROJ + SZ_PRE);

    k_pool<<<dim3(NP / 64, C_ / 64, B_), 256, 0, stream>>>(x, resid, xpbT);
    k_zero<<<1, 512, 0, stream>>>(stats);
    k_proj<<<dim3(NP / 64, 2, 3 * B_), 128, 0, stream>>>(thw, phw, gw, xpbT, thT, phiT, gC);
    k_attn<<<dim3(NP / 64, B_), 128, 0, stream>>>(thT, phiT, gC, ybuf);
    k_out<<<dim3(NP / 64, C_ / 64, B_), 128, 0, stream>>>(yw, ybuf, outpre, stats);
    k_bnrelu<<<(B_ * C_ * NN + 255) / 256, 256, 0, stream>>>(outpre, resid, stats, bnw, bnb, (float*)d_out);
}
